// MultiHeadXCITAttention_29033978921088
// MI455X (gfx1250) — matrix operand order repaired and checked by reading, not compiled, not run on hardware
//
#include <hip/hip_runtime.h>

// ---------------------------------------------------------------------------
// XCiT cross-covariance attention for MI455X (gfx1250, wave32, WMMA bf16)
// Round 2: software-pipelined k-loops (register double buffering) + wider
// wave tiles (32x64) in the big GEMMs; 2-way K-split energy kernel.
// ---------------------------------------------------------------------------

typedef __bf16 bf16;
typedef __attribute__((ext_vector_type(16))) __bf16 v16bf;
typedef __attribute__((ext_vector_type(8)))  __bf16 v8bf;
typedef __attribute__((ext_vector_type(4)))  __bf16 v4bf;
typedef __attribute__((ext_vector_type(8)))  float  v8f;

#define EMBED 768
#define HEADS 12
#define HDIM  64
#define SEQ   4096
#define BATCH 8

// ---- bf16 <-> f32 helpers (bit-level) -------------------------------------
__device__ __forceinline__ bf16 f2bf(float f) {
  union { float f; unsigned u; } c; c.f = f;
  unsigned u = c.u;
  unsigned r = (u + 0x7FFFu + ((u >> 16) & 1u)) >> 16;  // round-to-nearest-even
  union { unsigned short s; bf16 b; } o; o.s = (unsigned short)r;
  return o.b;
}
__device__ __forceinline__ float bf2f(bf16 b) {
  union { unsigned short s; bf16 b; } i; i.b = b;
  union { float f; unsigned u; } o; o.u = ((unsigned)i.s) << 16;
  return o.f;
}

// ---- WMMA fragment loads --------------------------------------------------
// A-fragment (16x32 bf16): lane = row (mod 16); contiguous 16-k chunk
__device__ __forceinline__ v16bf load_frag_a(const bf16* p) {
  v8bf lo = *(const v8bf*)p;
  v8bf hi = *(const v8bf*)(p + 8);
  v16bf r;
#pragma unroll
  for (int i = 0; i < 8; ++i) { r[i] = lo[i]; r[i + 8] = hi[i]; }
  return r;
}
// B-fragment (32x16 bf16): lane = col (mod 16); contiguous 16-k chunk
__device__ __forceinline__ v16bf load_frag_b(const bf16* p) {
  return *(const v16bf*)p;
}
__device__ __forceinline__ v8f wmma_bf16(v16bf a, v16bf b, v8f c) {
  return __builtin_amdgcn_wmma_f32_16x16x32_bf16(
      false, a, false, b, (short)0, c, false, false);
}

// ---------------------------------------------------------------------------
// fp32 -> bf16 conversion, 4 elements/thread
// ---------------------------------------------------------------------------
__global__ __launch_bounds__(256) void cvt_f32_bf16_x4(
    const float4* __restrict__ in, v4bf* __restrict__ out, int n4) {
  int i = blockIdx.x * 256 + threadIdx.x;
  if (i < n4) {
    float4 f = in[i];
    v4bf o;
    o[0] = f2bf(f.x); o[1] = f2bf(f.y); o[2] = f2bf(f.z); o[3] = f2bf(f.w);
    out[i] = o;
  }
}

// ---------------------------------------------------------------------------
// transpose + convert a 768x768 fp32 weight to bf16 [N,K]
// ---------------------------------------------------------------------------
__global__ __launch_bounds__(256) void wtr_kernel(
    const float* __restrict__ W, bf16* __restrict__ Wt) {
  int i = blockIdx.x * 256 + threadIdx.x;   // i = n*768 + k
  int n = i / EMBED, k = i - n * EMBED;
  Wt[i] = f2bf(W[(size_t)k * EMBED + n]);
}

// ---------------------------------------------------------------------------
// C[M,N] = A[M,K] @ BT[N,K]^T + bias ; WMMA bf16, f32 accumulate.
// 256 threads = 8 waves (4 in M x 2 in N); wave tile 32(M) x 64(N) =
// 2x4 WMMA tiles (8 WMMA per 6 fragment loads per 32-deep k-step).
// Workgroup tile 128 x 128.  K-loop is register double-buffered so the
// next step's global_load_b128s are in flight during the current WMMAs.
// ---------------------------------------------------------------------------
template <bool OUT_F32>
__global__ __launch_bounds__(256) void gemm_bf16(
    const bf16* __restrict__ A, const bf16* __restrict__ BT,
    const float* __restrict__ bias, void* __restrict__ C,
    int M, int N, int K) {
  const int tid  = threadIdx.x;
  const int wave = tid >> 5, lane = tid & 31;
  const int lhi  = lane >> 4, llo = lane & 15;
  const int wm   = wave >> 1, wn = wave & 1;
  const int m0   = blockIdx.x * 128 + wm * 32;
  const int n0   = blockIdx.y * 128 + wn * 64;

  const bf16* a0p = A + (size_t)(m0 + llo) * K + 16 * lhi;
  const bf16* a1p = a0p + (size_t)16 * K;
  const bf16* bp0 = BT + (size_t)(n0 + llo) * K + 16 * lhi;
  const bf16* bp1 = bp0 + (size_t)16 * K;
  const bf16* bp2 = bp1 + (size_t)16 * K;
  const bf16* bp3 = bp2 + (size_t)16 * K;

  v8f acc[2][4] = {};

  // prologue: fragments for k0 = 0
  v16bf a0 = load_frag_a(a0p), a1 = load_frag_a(a1p);
  v16bf b0 = load_frag_b(bp0), b1 = load_frag_b(bp1);
  v16bf b2 = load_frag_b(bp2), b3 = load_frag_b(bp3);

  for (int k0 = 0; k0 < K - 32; k0 += 32) {
    const int kn = k0 + 32;
    // issue next step's loads first (overlap with WMMAs below)
    v16bf a0n = load_frag_a(a0p + kn);
    v16bf a1n = load_frag_a(a1p + kn);
    v16bf b0n = load_frag_b(bp0 + kn);
    v16bf b1n = load_frag_b(bp1 + kn);
    v16bf b2n = load_frag_b(bp2 + kn);
    v16bf b3n = load_frag_b(bp3 + kn);

    acc[0][0] = wmma_bf16(a0, b0, acc[0][0]);
    acc[0][1] = wmma_bf16(a0, b1, acc[0][1]);
    acc[0][2] = wmma_bf16(a0, b2, acc[0][2]);
    acc[0][3] = wmma_bf16(a0, b3, acc[0][3]);
    acc[1][0] = wmma_bf16(a1, b0, acc[1][0]);
    acc[1][1] = wmma_bf16(a1, b1, acc[1][1]);
    acc[1][2] = wmma_bf16(a1, b2, acc[1][2]);
    acc[1][3] = wmma_bf16(a1, b3, acc[1][3]);

    a0 = a0n; a1 = a1n; b0 = b0n; b1 = b1n; b2 = b2n; b3 = b3n;
  }
  // epilogue
  acc[0][0] = wmma_bf16(a0, b0, acc[0][0]);
  acc[0][1] = wmma_bf16(a0, b1, acc[0][1]);
  acc[0][2] = wmma_bf16(a0, b2, acc[0][2]);
  acc[0][3] = wmma_bf16(a0, b3, acc[0][3]);
  acc[1][0] = wmma_bf16(a1, b0, acc[1][0]);
  acc[1][1] = wmma_bf16(a1, b1, acc[1][1]);
  acc[1][2] = wmma_bf16(a1, b2, acc[1][2]);
  acc[1][3] = wmma_bf16(a1, b3, acc[1][3]);

  float* Cf = (float*)C;
  bf16*  Cb = (bf16*)C;
#pragma unroll
  for (int mi = 0; mi < 2; ++mi)
#pragma unroll
    for (int ni = 0; ni < 4; ++ni) {
      int col = n0 + ni * 16 + llo;
      float bv = bias[col];
      int rb = m0 + mi * 16 + lhi * 8;
#pragma unroll
      for (int i = 0; i < 8; ++i) {
        float v = acc[mi][ni][i] + bv;
        size_t idx = (size_t)(rb + i) * N + col;
        if (OUT_F32) Cf[idx] = v; else Cb[idx] = f2bf(v);
      }
    }
}

// ---------------------------------------------------------------------------
// per-(batch,channel) 1/max(||.||_2, eps) over the token axis for q and k.
// Grid: B * (768/64) blocks; 256 threads = 64 channels x 4 token slices.
// ---------------------------------------------------------------------------
__global__ __launch_bounds__(256) void norm_kernel(
    const bf16* __restrict__ q, const bf16* __restrict__ k,
    float* __restrict__ invq, float* __restrict__ invk) {
  const int b  = blockIdx.x / (EMBED / 64);
  const int c0 = (blockIdx.x % (EMBED / 64)) * 64;
  const int c  = threadIdx.x & 63;
  const int s  = threadIdx.x >> 6;
  float aq = 0.f, ak = 0.f;
  for (int n = s; n < SEQ; n += 4) {
    size_t idx = ((size_t)b * SEQ + n) * EMBED + c0 + c;
    float x1 = bf2f(q[idx]); aq += x1 * x1;
    float x2 = bf2f(k[idx]); ak += x2 * x2;
  }
  __shared__ float Sq[256], Sk[256];
  Sq[threadIdx.x] = aq; Sk[threadIdx.x] = ak;
  __syncthreads();
  if (s == 0) {
    float tq = Sq[c] + Sq[c + 64] + Sq[c + 128] + Sq[c + 192];
    float tk = Sk[c] + Sk[c + 64] + Sk[c + 128] + Sk[c + 192];
    invq[(size_t)b * EMBED + c0 + c] = 1.f / fmaxf(sqrtf(tq), 1e-12f);
    invk[(size_t)b * EMBED + c0 + c] = 1.f / fmaxf(sqrtf(tk), 1e-12f);
  }
}

// ---------------------------------------------------------------------------
// normalize + transpose [B,N,C] -> [B,C,N] (bf16), 32x32 LDS tile transpose
// ---------------------------------------------------------------------------
__global__ __launch_bounds__(256) void ntr_kernel(
    const bf16* __restrict__ src, const float* __restrict__ inv,
    bf16* __restrict__ dstT) {
  const int b  = blockIdx.z;
  const int n0 = blockIdx.x * 32, c0 = blockIdx.y * 32;
  __shared__ float tile[32][33];
  const int tx = threadIdx.x & 31, ty = threadIdx.x >> 5;  // ty in [0,8)
#pragma unroll
  for (int j = 0; j < 32; j += 8) {
    int n = n0 + ty + j, c = c0 + tx;
    tile[ty + j][tx] =
        bf2f(src[((size_t)b * SEQ + n) * EMBED + c]) * inv[(size_t)b * EMBED + c];
  }
  __syncthreads();
#pragma unroll
  for (int j = 0; j < 32; j += 8) {
    int c = c0 + ty + j, n = n0 + tx;
    dstT[((size_t)b * EMBED + c) * SEQ + n] = f2bf(tile[tx][ty + j]);
  }
}

// ---------------------------------------------------------------------------
// fused energy GEMM + softmax.  One block of 256 threads per (b,h).
// energy[e,q] = sum_n qT[e,n]*kT[q,n]  (M=64, N=64, K=4096)
// 8 waves = 4 M-tile rows x 2 K-groups (each group sums half of K into its
// own padded LDS buffer); softmax over q adds the two partials, scales by
// 1/sqrt(E), stores transposed attnT[b,h,q,e] in bf16.
// ---------------------------------------------------------------------------
__global__ __launch_bounds__(256) void energy_softmax_kernel(
    const bf16* __restrict__ qT, const bf16* __restrict__ kT,
    bf16* __restrict__ attnT) {
  const int bh = blockIdx.x;
  const int b = bh / HEADS, h = bh % HEADS;
  const bf16* Q  = qT + ((size_t)b * EMBED + h * HDIM) * SEQ;
  const bf16* Kp = kT + ((size_t)b * EMBED + h * HDIM) * SEQ;

  const int tid  = threadIdx.x;
  const int wave = tid >> 5, lane = tid & 31;
  const int lhi  = lane >> 4, llo = lane & 15;
  const int kg   = wave >> 2;           // K group: 0 or 1
  const int m0   = (wave & 3) * 16;     // M tile row
  const int kbase = kg * (SEQ / 2);
  const int klen  = SEQ / 2;

  const bf16* ap  = Q + (size_t)(m0 + llo) * SEQ + kbase + 16 * lhi;
  const bf16* bq0 = Kp + (size_t)llo * SEQ + kbase + 16 * lhi;
  const bf16* bq1 = bq0 + (size_t)16 * SEQ;
  const bf16* bq2 = bq1 + (size_t)16 * SEQ;
  const bf16* bq3 = bq2 + (size_t)16 * SEQ;

  v8f acc[4] = {};
  v16bf a  = load_frag_a(ap);
  v16bf b0 = load_frag_b(bq0), b1 = load_frag_b(bq1);
  v16bf b2 = load_frag_b(bq2), b3 = load_frag_b(bq3);

  for (int k0 = 0; k0 < klen - 32; k0 += 32) {
    const int kn = k0 + 32;
    v16bf an  = load_frag_a(ap + kn);
    v16bf b0n = load_frag_b(bq0 + kn);
    v16bf b1n = load_frag_b(bq1 + kn);
    v16bf b2n = load_frag_b(bq2 + kn);
    v16bf b3n = load_frag_b(bq3 + kn);
    acc[0] = wmma_bf16(a, b0, acc[0]);
    acc[1] = wmma_bf16(a, b1, acc[1]);
    acc[2] = wmma_bf16(a, b2, acc[2]);
    acc[3] = wmma_bf16(a, b3, acc[3]);
    a = an; b0 = b0n; b1 = b1n; b2 = b2n; b3 = b3n;
  }
  acc[0] = wmma_bf16(a, b0, acc[0]);
  acc[1] = wmma_bf16(a, b1, acc[1]);
  acc[2] = wmma_bf16(a, b2, acc[2]);
  acc[3] = wmma_bf16(a, b3, acc[3]);

  __shared__ float Esm[2][64 * 65];   // +1 pad: conflict-free row reads
#pragma unroll
  for (int ni = 0; ni < 4; ++ni)
#pragma unroll
    for (int i = 0; i < 8; ++i)
      Esm[kg][(m0 + lhi * 8 + i) * 65 + ni * 16 + llo] = acc[ni][i];
  __syncthreads();

  if (tid < 64) {  // thread = row e; softmax over q (64)
    const float* r0 = &Esm[0][tid * 65];
    const float* r1 = &Esm[1][tid * 65];
    float mx = -3.4e38f;
    for (int qi = 0; qi < 64; ++qi) mx = fmaxf(mx, r0[qi] + r1[qi]);
    float s = 0.f;
    for (int qi = 0; qi < 64; ++qi) s += __expf(r0[qi] + r1[qi] - mx);
    float invs = 1.f / (s * sqrtf((float)EMBED));  // softmax / sqrt(E)
    bf16* outp = attnT + (size_t)bh * HDIM * HDIM; // [q][e]
    for (int qi = 0; qi < 64; ++qi)
      outp[(size_t)qi * HDIM + tid] = f2bf(__expf(r0[qi] + r1[qi] - mx) * invs);
  }
}

// ---------------------------------------------------------------------------
// Y = V @ attn per (b,h).  M=4096 tokens, N=64, K=64.
// Grid: (B*H, SEQ/128); 8 waves in 4x2; wave tile 32x32.  All 8 fragment
// loads for both k-steps are issued before the 8 WMMAs.
// ---------------------------------------------------------------------------
__global__ __launch_bounds__(256) void av_gemm(
    const bf16* __restrict__ V, const bf16* __restrict__ attnT,
    bf16* __restrict__ Y) {
  const int bh = blockIdx.x;
  const int b = bh / HEADS, h = bh % HEADS;
  const int tid  = threadIdx.x;
  const int wave = tid >> 5, lane = tid & 31;
  const int lhi  = lane >> 4, llo = lane & 15;
  const int wm   = wave >> 1, wn = wave & 1;
  const int m0   = blockIdx.y * 128 + wm * 32;  // token
  const int n0   = wn * 32;                     // out-channel within head

  const bf16* a0p = V + ((size_t)b * SEQ + m0 + llo) * EMBED + h * HDIM + 16 * lhi;
  const bf16* a1p = a0p + (size_t)16 * EMBED;
  const bf16* bb  = attnT + (size_t)bh * HDIM * HDIM;
  const bf16* b0p = bb + (size_t)(n0 + llo) * HDIM + 16 * lhi;
  const bf16* b1p = b0p + (size_t)16 * HDIM;

  // issue every load before any WMMA
  v16bf a0_0 = load_frag_a(a0p);
  v16bf a1_0 = load_frag_a(a1p);
  v16bf b0_0 = load_frag_b(b0p);
  v16bf b1_0 = load_frag_b(b1p);
  v16bf a0_1 = load_frag_a(a0p + 32);
  v16bf a1_1 = load_frag_a(a1p + 32);
  v16bf b0_1 = load_frag_b(b0p + 32);
  v16bf b1_1 = load_frag_b(b1p + 32);

  v8f acc[2][2] = {};
  acc[0][0] = wmma_bf16(a0_0, b0_0, acc[0][0]);
  acc[0][1] = wmma_bf16(a0_0, b1_0, acc[0][1]);
  acc[1][0] = wmma_bf16(a1_0, b0_0, acc[1][0]);
  acc[1][1] = wmma_bf16(a1_0, b1_0, acc[1][1]);
  acc[0][0] = wmma_bf16(a0_1, b0_1, acc[0][0]);
  acc[0][1] = wmma_bf16(a0_1, b1_1, acc[0][1]);
  acc[1][0] = wmma_bf16(a1_1, b0_1, acc[1][0]);
  acc[1][1] = wmma_bf16(a1_1, b1_1, acc[1][1]);

#pragma unroll
  for (int mi = 0; mi < 2; ++mi)
#pragma unroll
    for (int ni = 0; ni < 2; ++ni) {
      int col = n0 + ni * 16 + llo;
      int rb = m0 + mi * 16 + lhi * 8;
#pragma unroll
      for (int i = 0; i < 8; ++i)
        Y[((size_t)b * SEQ + rb + i) * EMBED + h * HDIM + col] =
            f2bf(acc[mi][ni][i]);
    }
}

// ---------------------------------------------------------------------------
// Launcher
// ---------------------------------------------------------------------------
extern "C" void kernel_launch(void* const* d_in, const int* in_sizes, int n_in,
                              void* d_out, int out_size, void* d_ws, size_t ws_size,
                              hipStream_t stream) {
  (void)in_sizes; (void)n_in; (void)out_size; (void)ws_size;

  const float* x  = (const float*)d_in[0];
  const float* Wq = (const float*)d_in[1];
  const float* bq = (const float*)d_in[2];
  const float* Wk = (const float*)d_in[3];
  const float* bk = (const float*)d_in[4];
  const float* Wv = (const float*)d_in[5];
  const float* bv = (const float*)d_in[6];
  const float* Wo = (const float*)d_in[7];
  const float* bo = (const float*)d_in[8];
  float* out = (float*)d_out;

  const size_t M  = (size_t)BATCH * SEQ;            // 32768
  const size_t XE = M * EMBED;                      // 25,165,824 elems
  const size_t WB = (size_t)EMBED * EMBED * sizeof(bf16);

  char* ws = (char*)d_ws;
  size_t off = 0;
  bf16* xb  = (bf16*)(ws + off); off += XE * sizeof(bf16);
  bf16* Wt0 = (bf16*)(ws + off); off += WB;   // WqT
  bf16* Wt1 = (bf16*)(ws + off); off += WB;   // WkT
  bf16* Wt2 = (bf16*)(ws + off); off += WB;   // WvT
  bf16* Wt3 = (bf16*)(ws + off); off += WB;   // WoT
  bf16* qb  = (bf16*)(ws + off); off += XE * sizeof(bf16);
  bf16* kb  = (bf16*)(ws + off); off += XE * sizeof(bf16);
  bf16* vb  = (bf16*)(ws + off); off += XE * sizeof(bf16);
  bf16* qT  = (bf16*)(ws + off); off += XE * sizeof(bf16);
  bf16* kT  = (bf16*)(ws + off); off += XE * sizeof(bf16);
  float* invq = (float*)(ws + off); off += (size_t)BATCH * EMBED * sizeof(float);
  float* invk = (float*)(ws + off); off += (size_t)BATCH * EMBED * sizeof(float);
  bf16* attnT = (bf16*)(ws + off); off += (size_t)BATCH * HEADS * HDIM * HDIM * sizeof(bf16);
  bf16* yb  = (bf16*)(ws + off); off += XE * sizeof(bf16);

  // 1) x -> bf16
  {
    int n4 = (int)(XE / 4);
    cvt_f32_bf16_x4<<<dim3((n4 + 255) / 256), dim3(256), 0, stream>>>(
        (const float4*)x, (v4bf*)xb, n4);
  }
  // 2) transpose+convert the four weights
  {
    int blocks = (EMBED * EMBED) / 256;  // 2304
    wtr_kernel<<<dim3(blocks), dim3(256), 0, stream>>>(Wq, Wt0);
    wtr_kernel<<<dim3(blocks), dim3(256), 0, stream>>>(Wk, Wt1);
    wtr_kernel<<<dim3(blocks), dim3(256), 0, stream>>>(Wv, Wt2);
    wtr_kernel<<<dim3(blocks), dim3(256), 0, stream>>>(Wo, Wt3);
  }
  // 3) QKV projections (bf16 out); workgroup tile 128x128
  {
    dim3 grid((unsigned)(M / 128), EMBED / 128);  // 256 x 6
    gemm_bf16<false><<<grid, dim3(256), 0, stream>>>(xb, Wt0, bq, (void*)qb,
                                                     (int)M, EMBED, EMBED);
    gemm_bf16<false><<<grid, dim3(256), 0, stream>>>(xb, Wt1, bk, (void*)kb,
                                                     (int)M, EMBED, EMBED);
    gemm_bf16<false><<<grid, dim3(256), 0, stream>>>(xb, Wt2, bv, (void*)vb,
                                                     (int)M, EMBED, EMBED);
  }
  // 4) per-channel token-axis L2 norms for q, k
  norm_kernel<<<dim3(BATCH * (EMBED / 64)), dim3(256), 0, stream>>>(
      qb, kb, invq, invk);
  // 5) normalize + transpose to [B, C, N]
  {
    dim3 grid(SEQ / 32, EMBED / 32, BATCH);  // 128 x 24 x 8
    ntr_kernel<<<grid, dim3(256), 0, stream>>>(qb, invq, qT);
    ntr_kernel<<<grid, dim3(256), 0, stream>>>(kb, invk, kT);
  }
  // 6) fused energy GEMM + softmax -> attnT (bf16, [b,h,q,e])
  energy_softmax_kernel<<<dim3(BATCH * HEADS), dim3(256), 0, stream>>>(
      qT, kT, attnT);
  // 7) Y = V @ attn (bf16)
  av_gemm<<<dim3(BATCH * HEADS, SEQ / 128), dim3(256), 0, stream>>>(
      vb, attnT, yb);
  // 8) output projection (fp32 out + bias)
  {
    dim3 grid((unsigned)(M / 128), EMBED / 128);
    gemm_bf16<true><<<grid, dim3(256), 0, stream>>>(yb, Wt3, bo, (void*)out,
                                                    (int)M, EMBED, EMBED);
  }
}
